// DYCE_25194278158366
// MI455X (gfx1250) — matrix operand order repaired and checked by reading, not compiled, not run on hardware
//
#include <hip/hip_runtime.h>
#include <hip/hip_bf16.h>

// ---------------------------------------------------------------------------
// Types for CDNA5 WMMA (wave32, 16x16x32 bf16 -> f32)
// ---------------------------------------------------------------------------
typedef __attribute__((ext_vector_type(16))) __bf16 v16bf;
typedef __attribute__((ext_vector_type(8)))  float  v8f;

union ABu {
    uint4          q[2];
    unsigned short s[16];
    v16bf          v;
};

#define MROWS (8 * 2048)   // BATCH * L = 16384
#define LSEQ  2048
#define EDIM  512
#define NST   16

__device__ __forceinline__ float gelu_f(float x) {
    float x3 = x * x * x;
    return 0.5f * x * (1.0f + tanhf(0.7978845608028654f * (x + 0.044715f * x3)));
}

// A tile (16x32 bf16), ISA 7.12.2 layout:
//   lanes 0-15 : elems 0-7 -> K=k0+0..7,  elems 8-15 -> K=k0+16..23
//   lanes16-31 : elems 0-7 -> K=k0+8..15, elems 8-15 -> K=k0+24..31
__device__ __forceinline__ v16bf load_a_vec(const __hip_bfloat16* __restrict__ rowp,
                                            int k0, int half) {
    ABu u;
    u.q[0] = *reinterpret_cast<const uint4*>(rowp + k0 + half * 8);
    u.q[1] = *reinterpret_cast<const uint4*>(rowp + k0 + 16 + half * 8);
    return u.v;
}

__device__ __forceinline__ v16bf load_a_tail(const __hip_bfloat16* __restrict__ rowp,
                                             int K, int k0, int half) {
    ABu u;
    for (int i = 0; i < 16; ++i) {
        int k = k0 + ((i < 8) ? (half * 8 + i) : (16 + half * 8 + (i - 8)));
        u.s[i] = (k < K) ? *reinterpret_cast<const unsigned short*>(rowp + k)
                         : (unsigned short)0;
    }
    return u.v;
}

// B tile (32x16 bf16), Bt stored N x K:
//   lanes 0-15 hold K=k0+0..7,16..23 of column n, lanes 16-31 hold K=k0+8..15,24..31
__device__ __forceinline__ v16bf load_b_vec(const __hip_bfloat16* __restrict__ colp,
                                            int k0, int half) {
    ABu u;
    u.q[0] = *reinterpret_cast<const uint4*>(colp + k0 + half * 8);
    u.q[1] = *reinterpret_cast<const uint4*>(colp + k0 + 16 + half * 8);
    return u.v;
}

__device__ __forceinline__ v16bf load_b_tail(const __hip_bfloat16* __restrict__ colp,
                                             int K, int k0, int half) {
    ABu u;
    for (int i = 0; i < 16; ++i) {
        int k = k0 + ((i < 8) ? (half * 8 + i) : (16 + half * 8 + (i - 8)));
        u.s[i] = (k < K) ? *reinterpret_cast<const unsigned short*>(colp + k)
                         : (unsigned short)0;
    }
    return u.v;
}

#define WMMA_BF16(a, b, c) \
    __builtin_amdgcn_wmma_f32_16x16x32_bf16(false, (a), false, (b), (short)0, (c), false, false)

// ---------------------------------------------------------------------------
// Register-blocked GEMM: each wave computes a 64x64 tile (4x4 WMMA tiles).
// 16 WMMAs per K-step with 16 b128 loads.  Requires N % 64 == 0, M % 64 == 0.
//   C = act( A[M,K] @ B[K,N] + bias + beta*C );  act: 0=none, 1=gelu
// ---------------------------------------------------------------------------
__global__ void __launch_bounds__(128)
gemm_wmma_bf16_blk(const __hip_bfloat16* __restrict__ A,
                   const __hip_bfloat16* __restrict__ Bt,
                   const float* __restrict__ bias,
                   float* __restrict__ C,
                   int M, int N, int K, int beta, int act) {
    int wave   = blockIdx.x * 4 + (threadIdx.x >> 5);
    int lane   = threadIdx.x & 31;
    int tilesN = N >> 6;
    int tm     = wave / tilesN;
    int tn     = wave - tm * tilesN;
    if (tm >= (M >> 6)) return;       // wave-uniform

    int half = lane >> 4;
    int lr   = lane & 15;

    const __hip_bfloat16* arow[4];
    const __hip_bfloat16* bcol[4];
#pragma unroll
    for (int i = 0; i < 4; ++i)
        arow[i] = A + (size_t)(tm * 64 + i * 16 + lr) * K;
#pragma unroll
    for (int j = 0; j < 4; ++j)
        bcol[j] = Bt + (size_t)(tn * 64 + j * 16 + lr) * K;

    v8f c[4][4];
#pragma unroll
    for (int i = 0; i < 4; ++i)
#pragma unroll
        for (int j = 0; j < 4; ++j) c[i][j] = v8f{};

    int kmain = K & ~31;
    for (int k0 = 0; k0 < kmain; k0 += 32) {
        v16bf a[4], b[4];
#pragma unroll
        for (int i = 0; i < 4; ++i) {
            __builtin_prefetch(arow[i] + k0 + 32, 0, 1);   // speculative, safe OOB
            a[i] = load_a_vec(arow[i], k0, half);
        }
#pragma unroll
        for (int j = 0; j < 4; ++j) {
            __builtin_prefetch(bcol[j] + k0 + 32, 0, 1);
            b[j] = load_b_vec(bcol[j], k0, half);
        }
#pragma unroll
        for (int i = 0; i < 4; ++i)
#pragma unroll
            for (int j = 0; j < 4; ++j)
                c[i][j] = WMMA_BF16(a[i], b[j], c[i][j]);
    }
    if (kmain < K) {                  // only K=16 GEMM hits this
        v16bf a[4], b[4];
#pragma unroll
        for (int i = 0; i < 4; ++i) a[i] = load_a_tail(arow[i], K, kmain, half);
#pragma unroll
        for (int j = 0; j < 4; ++j) b[j] = load_b_tail(bcol[j], K, kmain, half);
#pragma unroll
        for (int i = 0; i < 4; ++i)
#pragma unroll
            for (int j = 0; j < 4; ++j)
                c[i][j] = WMMA_BF16(a[i], b[j], c[i][j]);
    }

#pragma unroll
    for (int j = 0; j < 4; ++j) {
        int ncol   = tn * 64 + j * 16 + lr;
        float bval = bias ? bias[ncol] : 0.0f;
#pragma unroll
        for (int i = 0; i < 4; ++i) {
#pragma unroll
            for (int r = 0; r < 8; ++r) {
                int row    = tm * 64 + i * 16 + r + half * 8;
                size_t off = (size_t)row * N + ncol;
                float v = c[i][j][r] + bval;
                if (beta) v += C[off];
                if (act == 1) v = gelu_f(v);
                C[off] = v;
            }
        }
    }
}

// ---------------------------------------------------------------------------
// Simple GEMM (one 16x16 tile per wave) for N % 64 != 0 (the N=48 x-proj).
// ---------------------------------------------------------------------------
__global__ void gemm_wmma_bf16(const __hip_bfloat16* __restrict__ A,
                               const __hip_bfloat16* __restrict__ Bt,
                               const float* __restrict__ bias,
                               float* __restrict__ C,
                               int M, int N, int K, int beta, int act) {
    int wave   = blockIdx.x * (blockDim.x >> 5) + (threadIdx.x >> 5);
    int lane   = threadIdx.x & 31;
    int tilesN = N >> 4;
    int tm     = wave / tilesN;
    int tn     = wave - tm * tilesN;
    if (tm >= (M >> 4)) return;

    int half = lane >> 4;
    int lr   = lane & 15;
    const __hip_bfloat16* arow = A + (size_t)(tm * 16 + lr) * K;
    const __hip_bfloat16* bcol = Bt + (size_t)(tn * 16 + lr) * K;
    int n = tn * 16 + lr;

    v8f c = {};
    int kmain = K & ~31;
    for (int k0 = 0; k0 < kmain; k0 += 32) {
        __builtin_prefetch(arow + k0 + 32, 0, 1);
        __builtin_prefetch(bcol + k0 + 32, 0, 1);
        v16bf a = load_a_vec(arow, k0, half);
        v16bf b = load_b_vec(bcol, k0, half);
        c = WMMA_BF16(a, b, c);
    }
    if (kmain < K) {
        v16bf a = load_a_tail(arow, K, kmain, half);
        v16bf b = load_b_tail(bcol, K, kmain, half);
        c = WMMA_BF16(a, b, c);
    }

    float bval = bias ? bias[n] : 0.0f;
#pragma unroll
    for (int r = 0; r < 8; ++r) {
        int row    = tm * 16 + r + half * 8;
        size_t off = (size_t)row * N + n;
        float v = c[r] + bval;
        if (beta) v += C[off];
        if (act == 1) v = gelu_f(v);
        C[off] = v;
    }
}

// ---------------------------------------------------------------------------
// Elementwise / helper kernels
// ---------------------------------------------------------------------------
__global__ void to_bf16_kernel(const float* __restrict__ x,
                               __hip_bfloat16* __restrict__ y, int n) {
    int i = blockIdx.x * blockDim.x + threadIdx.x;
    if (i < n) y[i] = __float2bfloat16(x[i]);
}

// W[K][N] f32 -> Wt[N][K] bf16
__global__ void transpose_bf16_kernel(const float* __restrict__ W,
                                      __hip_bfloat16* __restrict__ Wt,
                                      int K, int N) {
    int i = blockIdx.x * blockDim.x + threadIdx.x;
    if (i >= K * N) return;
    int k = i / N, n = i - k * N;
    Wt[(size_t)n * K + k] = __float2bfloat16(W[i]);
}

// RMSNorm over rows of 256 + convert to bf16 (blockDim must be 256)
__global__ void rmsnorm_bf16_kernel(const float* __restrict__ x,
                                    const float* __restrict__ w,
                                    __hip_bfloat16* __restrict__ out) {
    __shared__ float red[256];
    int row = blockIdx.x;
    int i   = threadIdx.x;
    float v = x[(size_t)row * 256 + i];
    red[i] = v * v;
    __syncthreads();
    for (int s = 128; s > 0; s >>= 1) {
        if (i < s) red[i] += red[i + s];
        __syncthreads();
    }
    float scale = rsqrtf(red[0] * (1.0f / 256.0f) + 1e-5f);
    out[(size_t)row * 256 + i] = __float2bfloat16(v * scale * w[i]);
}

// Depthwise causal conv (DC=4) + SiLU on xc half of xz; writes f32 and bf16.
__global__ void conv_silu_kernel(const float* __restrict__ xz,
                                 const float* __restrict__ cw,   // [ED][4]
                                 const float* __restrict__ cb,   // [ED]
                                 float* __restrict__ xcf,
                                 __hip_bfloat16* __restrict__ xcb,
                                 int total) {
    int idx = blockIdx.x * blockDim.x + threadIdx.x;
    if (idx >= total) return;
    int e   = idx & (EDIM - 1);
    int row = idx >> 9;                 // b*L + t
    int t   = row & (LSEQ - 1);
    float acc = cb[e];
#pragma unroll
    for (int k = 0; k < 4; ++k) {
        int tt = t - 3 + k;
        if (tt >= 0)
            acc += xz[(size_t)(row + tt - t) * 1024 + e] * cw[e * 4 + k];
    }
    float s = acc / (1.0f + expf(-acc));   // silu
    xcf[idx] = s;
    xcb[idx] = __float2bfloat16(s);
}

// dbc[:, :16] -> bf16 dlt
__global__ void extract_dlt_kernel(const float* __restrict__ dbc,
                                   __hip_bfloat16* __restrict__ dlt, int M) {
    int idx = blockIdx.x * blockDim.x + threadIdx.x;
    if (idx >= M * 16) return;
    int row = idx >> 4, j = idx & 15;
    dlt[idx] = __float2bfloat16(dbc[(size_t)row * 48 + j]);
}

// Selective scan: 16 lanes per (batch, channel) -- one lane per state n.
// Fuses softplus(delta), A=-exp(A_log), D*u skip, SiLU(zg) gate.
// y-reduction over the 16-lane group via __shfl_xor (wave32 cross-lane).
__global__ void scan_kernel(const float* __restrict__ xcf,
                            const float* __restrict__ draw,
                            const float* __restrict__ dbc,
                            const float* __restrict__ xz,
                            const float* __restrict__ A_log_l,   // [ED][16]
                            const float* __restrict__ D_l,       // [ED]
                            __hip_bfloat16* __restrict__ yb) {
    int idx  = blockIdx.x * blockDim.x + threadIdx.x;
    int pair = idx >> 4;            // (b, e)
    int n    = idx & 15;            // state index
    if (pair >= 8 * EDIM) return;
    int b = pair >> 9;
    int e = pair & (EDIM - 1);

    float Ac = -expf(A_log_l[e * NST + n]);
    float hn = 0.0f;
    float Dp = D_l[e];

    for (int t = 0; t < LSEQ; ++t) {
        size_t row = (size_t)b * LSEQ + t;
        float dr    = draw[row * EDIM + e];
        float delta = (dr > 20.0f) ? dr : log1pf(expf(dr));   // softplus
        float u     = xcf[row * EDIM + e];
        float du    = delta * u;
        float Bv    = dbc[row * 48 + 16 + n];
        float Cv    = dbc[row * 48 + 32 + n];
        hn = expf(delta * Ac) * hn + du * Bv;
        float y = hn * Cv;
#pragma unroll
        for (int o = 8; o >= 1; o >>= 1) y += __shfl_xor(y, o, 16);
        if (n == 0) {
            y += Dp * u;
            float zg = xz[row * 1024 + EDIM + e];
            float g  = zg / (1.0f + expf(-zg));               // silu
            yb[row * EDIM + e] = __float2bfloat16(y * g);
        }
    }
}

// Final tiny head: gelu(h1 @ W_h2 + b_h2), N=2 -> scalar
__global__ void head_kernel(const float* __restrict__ h1,
                            const float* __restrict__ Wh2,
                            const float* __restrict__ bh2,
                            float* __restrict__ out, int M) {
    int row = blockIdx.x * blockDim.x + threadIdx.x;
    if (row >= M) return;
    float a0 = bh2[0], a1 = bh2[1];
    for (int j = 0; j < 64; ++j) {
        float v = h1[(size_t)row * 64 + j];
        a0 += v * Wh2[j * 2 + 0];
        a1 += v * Wh2[j * 2 + 1];
    }
    out[row * 2 + 0] = gelu_f(a0);
    out[row * 2 + 1] = gelu_f(a1);
}

// ---------------------------------------------------------------------------
// Host launcher
// ---------------------------------------------------------------------------
static inline void launch_gemm64(const __hip_bfloat16* A, const __hip_bfloat16* Bt,
                                 const float* bias, float* C,
                                 int M, int N, int K, int beta, int act,
                                 hipStream_t stream) {
    int tiles = (M >> 6) * (N >> 6);         // 64x64 tiles, one per wave
    gemm_wmma_bf16_blk<<<tiles / 4, 128, 0, stream>>>(A, Bt, bias, C, M, N, K, beta, act);
}

static inline void launch_gemm16(const __hip_bfloat16* A, const __hip_bfloat16* Bt,
                                 const float* bias, float* C,
                                 int M, int N, int K, int beta, int act,
                                 hipStream_t stream) {
    int tiles = (M >> 4) * (N >> 4);
    gemm_wmma_bf16<<<tiles / 4, 128, 0, stream>>>(A, Bt, bias, C, M, N, K, beta, act);
}

extern "C" void kernel_launch(void* const* d_in, const int* in_sizes, int n_in,
                              void* d_out, int out_size, void* d_ws, size_t ws_size,
                              hipStream_t stream) {
    (void)in_sizes; (void)n_in; (void)out_size; (void)ws_size;
    const float* x        = (const float*)d_in[0];
    const float* Wsp      = (const float*)d_in[1];
    const float* Wfc      = (const float*)d_in[2];
    const float* b_fc     = (const float*)d_in[3];
    const float* norm_w   = (const float*)d_in[4];
    const float* in_proj  = (const float*)d_in[5];
    const float* conv_w   = (const float*)d_in[6];
    const float* conv_b   = (const float*)d_in[7];
    const float* x_proj   = (const float*)d_in[8];
    const float* dt_w     = (const float*)d_in[9];
    const float* dt_b     = (const float*)d_in[10];
    const float* A_log    = (const float*)d_in[11];
    const float* D_param  = (const float*)d_in[12];
    const float* out_proj = (const float*)d_in[13];
    const float* norm_f_w = (const float*)d_in[14];
    const float* W_h1     = (const float*)d_in[15];
    const float* b_h1     = (const float*)d_in[16];
    const float* W_h2     = (const float*)d_in[17];
    const float* b_h2     = (const float*)d_in[18];
    float* out = (float*)d_out;

    const int M = MROWS;   // 16384

    char* ws  = (char*)d_ws;
    size_t off = 0;
    auto alloc = [&](size_t bytes) -> char* {
        char* p = ws + off;
        off += (bytes + 255) & ~(size_t)255;
        return p;
    };

    __hip_bfloat16* act_bf = (__hip_bfloat16*)alloc((size_t)M * 1024 * 2);
    __hip_bfloat16* wt     = (__hip_bfloat16*)alloc((size_t)1024 * 1024 * 2);
    float*          z1     = (float*)alloc((size_t)M * 768 * 4);
    float*          h      = (float*)alloc((size_t)M * 256 * 4);
    __hip_bfloat16* xin_bf = (__hip_bfloat16*)alloc((size_t)M * 256 * 2);
    float*          xz     = (float*)alloc((size_t)M * 1024 * 4);
    float*          xcf    = (float*)alloc((size_t)M * 512 * 4);
    __hip_bfloat16* xcb    = (__hip_bfloat16*)alloc((size_t)M * 512 * 2);
    float*          dbc    = (float*)alloc((size_t)M * 48 * 4);
    __hip_bfloat16* dltb   = (__hip_bfloat16*)alloc((size_t)M * 16 * 2);
    float*          draw   = (float*)alloc((size_t)M * 512 * 4);
    __hip_bfloat16* yb     = (__hip_bfloat16*)alloc((size_t)M * 512 * 2);
    float*          h1     = (float*)alloc((size_t)M * 64 * 4);

    const int T = 256;
    auto nb = [](int n, int t) { return (n + t - 1) / t; };

    // Stage 1: z1 = x @ W_spatial    (M x 768, K=1024)
    to_bf16_kernel<<<nb(M * 1024, T), T, 0, stream>>>(x, act_bf, M * 1024);
    transpose_bf16_kernel<<<nb(1024 * 768, T), T, 0, stream>>>(Wsp, wt, 1024, 768);
    launch_gemm64(act_bf, wt, nullptr, z1, M, 768, 1024, 0, 0, stream);

    // Stage 2: h = z1 @ W_fc + b_fc  (M x 256, K=768)
    to_bf16_kernel<<<nb(M * 768, T), T, 0, stream>>>(z1, act_bf, M * 768);
    transpose_bf16_kernel<<<nb(768 * 256, T), T, 0, stream>>>(Wfc, wt, 768, 256);
    launch_gemm64(act_bf, wt, b_fc, h, M, 256, 768, 0, 0, stream);

    // Mamba layers
    for (int l = 0; l < 6; ++l) {
        rmsnorm_bf16_kernel<<<M, 256, 0, stream>>>(h, norm_w + l * 256, xin_bf);

        transpose_bf16_kernel<<<nb(256 * 1024, T), T, 0, stream>>>(
            in_proj + (size_t)l * 256 * 1024, wt, 256, 1024);
        launch_gemm64(xin_bf, wt, nullptr, xz, M, 1024, 256, 0, 0, stream);

        conv_silu_kernel<<<nb(M * 512, T), T, 0, stream>>>(
            xz, conv_w + (size_t)l * 512 * 4, conv_b + (size_t)l * 512,
            xcf, xcb, M * 512);

        transpose_bf16_kernel<<<nb(512 * 48, T), T, 0, stream>>>(
            x_proj + (size_t)l * 512 * 48, wt, 512, 48);
        launch_gemm16(xcb, wt, nullptr, dbc, M, 48, 512, 0, 0, stream);

        extract_dlt_kernel<<<nb(M * 16, T), T, 0, stream>>>(dbc, dltb, M);

        transpose_bf16_kernel<<<nb(16 * 512, T), T, 0, stream>>>(
            dt_w + (size_t)l * 16 * 512, wt, 16, 512);
        launch_gemm64(dltb, wt, dt_b + (size_t)l * 512, draw, M, 512, 16, 0, 0, stream);

        scan_kernel<<<nb(8 * EDIM * 16, T), T, 0, stream>>>(
            xcf, draw, dbc, xz,
            A_log + (size_t)l * 512 * 16, D_param + (size_t)l * 512, yb);

        transpose_bf16_kernel<<<nb(512 * 256, T), T, 0, stream>>>(
            out_proj + (size_t)l * 512 * 256, wt, 512, 256);
        launch_gemm64(yb, wt, nullptr, h, M, 256, 512, /*beta=*/1, 0, stream);
    }

    // Head
    rmsnorm_bf16_kernel<<<M, 256, 0, stream>>>(h, norm_f_w, xin_bf);
    transpose_bf16_kernel<<<nb(256 * 64, T), T, 0, stream>>>(W_h1, wt, 256, 64);
    launch_gemm64(xin_bf, wt, b_h1, h1, M, 64, 256, 0, /*gelu=*/1, stream);
    head_kernel<<<nb(M, T), T, 0, stream>>>(h1, W_h2, b_h2, out, M);
}
